// ConditionPooler_82300163326268
// MI455X (gfx1250) — matrix operand order repaired and checked by reading, not compiled, not run on hardware
//
#include <hip/hip_runtime.h>
#include <hip/hip_bf16.h>
#include <math.h>

#define C  512
#define T  32
#define H  8
#define DH 64

typedef __attribute__((ext_vector_type(16))) _Float16 v16h;
typedef __attribute__((ext_vector_type(8)))  _Float16 v8h;
typedef __attribute__((ext_vector_type(8)))  float    v8f;
typedef __attribute__((ext_vector_type(4)))  unsigned int v4u;
typedef __attribute__((ext_vector_type(8)))  int      v8i;
typedef __attribute__((ext_vector_type(4)))  int      v4i;

// ---------------------------------------------------------------------------
// WMMA fragment helpers (contiguous k order k = 16h + s, same for A and B; wave32)
// ---------------------------------------------------------------------------

// A-matrix 16x32 f16 from row-major [row][k] storage with leading dim `ld`
// lane<16 : M=lane,    K = 0..15;  lane>=16: M=lane-16, K = 16..31
__device__ __forceinline__ v16h load_a_frag(const _Float16* base, int ld) {
  const int lane = threadIdx.x & 31;
  const int row  = lane & 15;
  const int ko   = (lane < 16) ? 0 : 16;
  const _Float16* p = base + row * ld + ko;
  v8h lo = *(const v8h*)(p);
  v8h hi = *(const v8h*)(p + 8);
  v16h r;
#pragma unroll
  for (int i = 0; i < 8; ++i) { r[i] = lo[i]; r[i + 8] = hi[i]; }
  return r;
}

// B-matrix 32x16 f16: per-lane 16 consecutive K halves of one column.
__device__ __forceinline__ v16h load_b16(const _Float16* p) {
  v8h lo = *(const v8h*)(p);
  v8h hi = *(const v8h*)(p + 8);
  v16h r;
#pragma unroll
  for (int i = 0; i < 8; ++i) { r[i] = lo[i]; r[i + 8] = hi[i]; }
  return r;
}

// B-fragment from [col][k] storage (ld): lane<16 -> col=lane, K 0..15;
// lane>=16 -> col=lane-16, K 16..31.
__device__ __forceinline__ v16h load_b_frag(const _Float16* base, int ld) {
  const int lane = threadIdx.x & 31;
  const int col  = lane & 15;
  const int ko   = (lane < 16) ? 0 : 16;
  return load_b16(base + col * ld + ko);
}

__device__ __forceinline__ v8f wmma_f16(v16h a, v16h b, v8f c) {
  return __builtin_amdgcn_wmma_f32_16x16x32_f16(false, a, false, b,
                                                (short)0, c, false, false);
}

__device__ __forceinline__ v8f vzero8() {
  v8f z;
#pragma unroll
  for (int i = 0; i < 8; ++i) z[i] = 0.f;
  return z;
}

// ---------------------------------------------------------------------------
// Tensor Data Mover: async DMA of a 128x32 f16 tile (row-major, row stride
// `ld` elems) from global into LDS. D# layout per CDNA5 ISA.
// Issue from ONE wave; completion via s_wait_tensorcnt.
// ---------------------------------------------------------------------------
__device__ __forceinline__ void tdm_load_128x32(_Float16* lds_dst,
                                                const _Float16* gsrc, int ld) {
  const unsigned long long ga = (unsigned long long)(const void*)gsrc;
  const unsigned int laddr = (unsigned int)(unsigned long long)(void*)lds_dst;

  v4u g0;
  g0[0] = 1u;                                            // count=1, user mode
  g0[1] = laddr;                                         // lds_addr (bytes)
  g0[2] = (unsigned int)ga;                              // global_addr[31:0]
  g0[3] = ((unsigned int)(ga >> 32) & 0x01FFFFFFu)       // global_addr[56:32]
          | (2u << 30);                                  // type = 2 ("image")

  v8i g1;
  g1[0] = (1 << 16);          // workgroup_mask=0, data_size=1 (2 bytes)
  g1[1] = (32 << 16);         // atomic_barrier_addr=0 | tensor_dim0[15:0]=32
  g1[2] = (128 << 16);        // tensor_dim0[31:16]=0  | tensor_dim1[15:0]=128
  g1[3] = (32 << 16);         // tensor_dim1[31:16]=0  | tile_dim0=32
  g1[4] = 128;                // tile_dim1=128, tile_dim2=0
  g1[5] = ld;                 // tensor_dim0_stride[31:0] (elements)
  g1[6] = 0;                  // stride[47:32]=0 | tensor_dim1_stride lo=0
  g1[7] = 0;

  v4i z4 = {0, 0, 0, 0};
#if __clang_major__ >= 23
  v8i z8 = {0, 0, 0, 0, 0, 0, 0, 0};
  __builtin_amdgcn_tensor_load_to_lds(g0, g1, z4, z4, z8, 0);
#else
  __builtin_amdgcn_tensor_load_to_lds(g0, g1, z4, z4, 0);
#endif
}

// ---------------------------------------------------------------------------
// Shared 128x128 GEMM tile core: acc[jt] += A(128xK) @ W(128xK)^T tile
// TDM double-buffered LDS; 8 waves, wave w owns rows 16w..16w+15.
// All 8 B-fragments are staged before the WMMA chain so the scheduler can
// issue the ds_loads in bulk and drain WMMAs behind partial s_wait_dscnt.
// ---------------------------------------------------------------------------
__device__ __forceinline__ void gemm_core_128x128(
    const _Float16* __restrict__ Abase, const _Float16* __restrict__ Wbase,
    int K, _Float16* AlBase, _Float16* BlBase, v8f acc[8]) {
  const int wv = threadIdx.x >> 5;
  const bool issuer = (threadIdx.x < 32);
  const int iters = K >> 5;

  if (issuer) {                       // preload chunk 0 into buffer 0
    tdm_load_128x32(AlBase, Abase, K);
    tdm_load_128x32(BlBase, Wbase, K);
  }
  for (int i = 0; i < iters; ++i) {
    const int buf = i & 1;
    if (issuer) {
      if (i + 1 < iters) {            // prefetch next chunk into other buffer
        tdm_load_128x32(AlBase + (buf ^ 1) * 128 * 32, Abase + (i + 1) * 32, K);
        tdm_load_128x32(BlBase + (buf ^ 1) * 128 * 32, Wbase + (i + 1) * 32, K);
        __builtin_amdgcn_s_wait_tensorcnt(2);  // in-order: oldest pair done
      } else {
        __builtin_amdgcn_s_wait_tensorcnt(0);
      }
    }
    __syncthreads();                  // publish LDS tile to all waves
    const _Float16* Al = AlBase + buf * 128 * 32;
    const _Float16* Bl = BlBase + buf * 128 * 32;
    const v16h a = load_a_frag(Al + wv * 16 * 32, 32);
    v16h bfr[8];
#pragma unroll
    for (int jt = 0; jt < 8; ++jt)
      bfr[jt] = load_b_frag(Bl + jt * 16 * 32, 32);
#pragma unroll
    for (int jt = 0; jt < 8; ++jt)
      acc[jt] = wmma_f16(a, bfr[jt], acc[jt]);
    __syncthreads();                  // protect buffer reuse 2 iters later
  }
}

// ---------------------------------------------------------------------------
// LayerNorm over rows of width C (output fp32 or f16)
// ---------------------------------------------------------------------------
template <typename OutT>
__global__ void ln_rows_kernel(const float* __restrict__ x,
                               const float* __restrict__ g,
                               const float* __restrict__ bta,
                               OutT* __restrict__ y) {
  __shared__ float red[2][8];
  const int row = blockIdx.x;
  const float* xr = x + (size_t)row * C;
  float s = 0.f, s2 = 0.f;
  for (int c = threadIdx.x; c < C; c += blockDim.x) {
    float v = xr[c]; s += v; s2 += v * v;
  }
#pragma unroll
  for (int off = 1; off < 32; off <<= 1) {
    s  += __shfl_xor(s,  off, 32);
    s2 += __shfl_xor(s2, off, 32);
  }
  const int w  = threadIdx.x >> 5;
  const int nw = blockDim.x >> 5;
  if ((threadIdx.x & 31) == 0) { red[0][w] = s; red[1][w] = s2; }
  __syncthreads();
  s = 0.f; s2 = 0.f;
  for (int i = 0; i < nw; ++i) { s += red[0][i]; s2 += red[1][i]; }
  const float mu  = s / (float)C;
  const float var = s2 / (float)C - mu * mu;
  const float inv = rsqrtf(var + 1e-5f);
  for (int c = threadIdx.x; c < C; c += blockDim.x)
    y[(size_t)row * C + c] = (OutT)((xr[c] - mu) * inv * g[c] + bta[c]);
}

// ---------------------------------------------------------------------------
// Q projection (tiny): qh16[h][t][d] = f16(((q_ln @ w_q^T) + b) / sqrt(DH))
// ---------------------------------------------------------------------------
__global__ void qproj_kernel(const float* __restrict__ q_ln,
                             const float* __restrict__ w_q,
                             const float* __restrict__ b_q_in,
                             _Float16* __restrict__ qh16) {
  const int idx = blockIdx.x * blockDim.x + threadIdx.x;  // T*C total
  const int t = idx / C, j = idx % C;
  const float* qr = q_ln + (size_t)t * C;
  const float* wr = w_q + (size_t)j * C;
  float acc = 0.f;
  for (int k = 0; k < C; ++k) acc += qr[k] * wr[k];
  acc = (acc + b_q_in[j]) * 0.125f;  // 1/sqrt(64)
  const int h = j >> 6, d = j & 63;
  qh16[((size_t)h * T + t) * DH + d] = (_Float16)acc;
}

__global__ void f32_to_f16_kernel(const float* __restrict__ src,
                                  _Float16* __restrict__ dst, size_t n) {
  size_t i = (size_t)blockIdx.x * blockDim.x + threadIdx.x;
  if (i < n) dst[i] = (_Float16)src[i];
}

// ---------------------------------------------------------------------------
// K/V projection: out[n][j] = feat[n][:] . w[j][:] + bias[j]
// z=0 -> K row-major [n][C]; z=1 -> V transposed [C][n]
// ---------------------------------------------------------------------------
__global__ void kvproj_kernel(const _Float16* __restrict__ feat16,
                              const _Float16* __restrict__ w16k,
                              const _Float16* __restrict__ w16v,
                              const float* __restrict__ bk,
                              const float* __restrict__ bv,
                              _Float16* __restrict__ K16,
                              _Float16* __restrict__ Vt16,
                              int N) {
  __shared__ __align__(16) _Float16 Al[2][128 * 32];
  __shared__ __align__(16) _Float16 Bl[2][128 * 32];
  const int which = blockIdx.z;
  const _Float16* w16  = which ? w16v : w16k;
  const float*    bias = which ? bv   : bk;
  const int m0 = blockIdx.x * 128;
  const int j0 = blockIdx.y * 128;
  const int wv = threadIdx.x >> 5;
  const int lane = threadIdx.x & 31;

  v8f acc[8];
#pragma unroll
  for (int i = 0; i < 8; ++i) acc[i] = vzero8();

  gemm_core_128x128(feat16 + (size_t)m0 * C, w16 + (size_t)j0 * C, C,
                    &Al[0][0], &Bl[0][0], acc);

  const int half = (lane >= 16) ? 8 : 0;
  if (which == 0) {
#pragma unroll
    for (int jt = 0; jt < 8; ++jt) {
      const int j = j0 + jt * 16 + (lane & 15);
      const float bj = bias[j];
#pragma unroll
      for (int r = 0; r < 8; ++r) {
        const int n = m0 + wv * 16 + half + r;
        K16[(size_t)n * C + j] = (_Float16)(acc[jt][r] + bj);
      }
    }
  } else {
#pragma unroll
    for (int jt = 0; jt < 8; ++jt) {
      const int j = j0 + jt * 16 + (lane & 15);
      const float bj = bias[j];
      v8h pack;
#pragma unroll
      for (int r = 0; r < 8; ++r) pack[r] = (_Float16)(acc[jt][r] + bj);
      const int n = m0 + wv * 16 + half;
      *(v8h*)&Vt16[(size_t)j * N + n] = pack;   // 8 consecutive n -> one b128
    }
  }
}

// ---------------------------------------------------------------------------
// Per-batch segment bounds (batch_idx sorted)
// ---------------------------------------------------------------------------
__global__ void seg_init_kernel(int* segs) {
  if (threadIdx.x < 32) segs[threadIdx.x] = 0;
}
__global__ void seg_bounds_kernel(const int* __restrict__ bidx,
                                  int* __restrict__ segs, int N) {
  const int i = blockIdx.x * blockDim.x + threadIdx.x;
  if (i >= N) return;
  const int b = bidx[i];
  if (i == 0) segs[b] = 0;
  else {
    const int p = bidx[i - 1];
    if (p != b) { segs[b] = i; segs[16 + p] = i; }
  }
  if (i == N - 1) segs[16 + b] = N;
}

// ---------------------------------------------------------------------------
// Flash attention: block = (h, b), 2 waves x 16 query rows, 32-key chunks.
// Scores + P*V on WMMA; online softmax in D-layout; f16 output.
// ---------------------------------------------------------------------------
__global__ void attn_kernel(const _Float16* __restrict__ qh16,  // [H][T][DH]
                            const _Float16* __restrict__ K16,   // [N][C]
                            const _Float16* __restrict__ Vt16,  // [C][N]
                            const int* __restrict__ segs,
                            _Float16* __restrict__ attnO16,     // [B][T][C]
                            int N) {
  __shared__ __align__(16) _Float16 plds[2][16 * 32];
  const int h  = blockIdx.x;
  const int b  = blockIdx.y;
  const int wv = threadIdx.x >> 5;
  const int lane = threadIdx.x & 31;
  const int tb = wv * 16;
  const int s0 = segs[b], s1 = segs[16 + b];

  const _Float16* qbase = qh16 + ((size_t)h * T + tb) * DH;
  const v16h aq0 = load_a_frag(qbase, DH);        // d 0..31
  const v16h aq1 = load_a_frag(qbase + 32, DH);   // d 32..63

  v8f o[4];
#pragma unroll
  for (int dt = 0; dt < 4; ++dt) o[dt] = vzero8();
  float rm[8], rs[8];
#pragma unroll
  for (int r = 0; r < 8; ++r) { rm[r] = -1e30f; rs[r] = 0.f; }

  const int ko   = (lane < 16) ? 0 : 16;
  const int half = (lane >= 16) ? 8 : 0;
  const _Float16* vtbase = Vt16 + (size_t)(h * DH + (lane & 15)) * N + ko;

  if (s0 < s1) {
    const int cbase0 = s0 & ~31;
    for (int cb = cbase0; cb < s1; cb += 32) {
      const int key0 = cb + (lane & 15);
      const int key1 = key0 + 16;
      const _Float16* kb0 = K16 + (size_t)key0 * C + h * DH;
      const _Float16* kb1 = K16 + (size_t)key1 * C + h * DH;
      const v16h bk00 = load_b16(kb0 + ko);
      const v16h bk01 = load_b16(kb0 + 32 + ko);
      const v16h bk10 = load_b16(kb1 + ko);
      const v16h bk11 = load_b16(kb1 + 32 + ko);

      v8f sc0 = vzero8(), sc1 = vzero8();
      sc0 = wmma_f16(aq0, bk00, sc0);
      sc0 = wmma_f16(aq1, bk01, sc0);
      sc1 = wmma_f16(aq0, bk10, sc1);
      sc1 = wmma_f16(aq1, bk11, sc1);

      if (key0 < s0 || key0 >= s1) {
#pragma unroll
        for (int r = 0; r < 8; ++r) sc0[r] = -1e30f;
      }
      if (key1 < s0 || key1 >= s1) {
#pragma unroll
        for (int r = 0; r < 8; ++r) sc1[r] = -1e30f;
      }

      float nm[8], cf[8];
#pragma unroll
      for (int r = 0; r < 8; ++r) {
        float cm = fmaxf(sc0[r], sc1[r]);
#pragma unroll
        for (int off = 1; off < 16; off <<= 1)
          cm = fmaxf(cm, __shfl_xor(cm, off, 32));
        nm[r] = fmaxf(rm[r], cm);
        cf[r] = __expf(rm[r] - nm[r]);
        rm[r] = nm[r];
      }
#pragma unroll
      for (int dt = 0; dt < 4; ++dt)
#pragma unroll
        for (int r = 0; r < 8; ++r) o[dt][r] *= cf[r];

      float p0[8], p1[8];
#pragma unroll
      for (int r = 0; r < 8; ++r) {
        p0[r] = __expf(sc0[r] - nm[r]);
        p1[r] = __expf(sc1[r] - nm[r]);
        float cs = p0[r] + p1[r];
#pragma unroll
        for (int off = 1; off < 16; off <<= 1)
          cs += __shfl_xor(cs, off, 32);
        rs[r] = rs[r] * cf[r] + cs;
      }

      // transpose P through LDS into an A-fragment
#pragma unroll
      for (int r = 0; r < 8; ++r) {
        plds[wv][(half + r) * 32 + (lane & 15)]      = (_Float16)p0[r];
        plds[wv][(half + r) * 32 + 16 + (lane & 15)] = (_Float16)p1[r];
      }
      __syncthreads();
      const v16h pa = load_a_frag(&plds[wv][0], 32);

      v16h bvv[4];
#pragma unroll
      for (int dt = 0; dt < 4; ++dt)
        bvv[dt] = load_b16(vtbase + (size_t)(dt * 16) * N + cb);
#pragma unroll
      for (int dt = 0; dt < 4; ++dt)
        o[dt] = wmma_f16(pa, bvv[dt], o[dt]);
      __syncthreads();
    }
  }

#pragma unroll
  for (int r = 0; r < 8; ++r) {
    const float inv = (rs[r] > 0.f) ? (1.f / rs[r]) : 0.f;
    const int t = tb + half + r;
#pragma unroll
    for (int dt = 0; dt < 4; ++dt)
      attnO16[((size_t)b * T + t) * C + h * DH + dt * 16 + (lane & 15)] =
          (_Float16)(o[dt][r] * inv);
  }
}

// ---------------------------------------------------------------------------
// Epilogue GEMMs on WMMA: out = A(MxK) @ W(NcolsxK)^T + bias, fused epilogue.
// MODE 0: + query residual -> f32   (w_o)
// MODE 1: exact GELU       -> f16   (ffn up)
// MODE 2: + hbuf residual  -> f32   (ffn down)
// ---------------------------------------------------------------------------
template <int MODE>
__global__ void wmma_gemm_kernel(const _Float16* __restrict__ A16,
                                 const _Float16* __restrict__ W16,
                                 const float* __restrict__ bias,
                                 const float* __restrict__ extra,
                                 float* __restrict__ outf,
                                 _Float16* __restrict__ outh,
                                 int K, int Ncols) {
  __shared__ __align__(16) _Float16 Al[2][128 * 32];
  __shared__ __align__(16) _Float16 Bl[2][128 * 32];
  const int m0 = blockIdx.x * 128;
  const int j0 = blockIdx.y * 128;
  const int wv = threadIdx.x >> 5;
  const int lane = threadIdx.x & 31;

  v8f acc[8];
#pragma unroll
  for (int i = 0; i < 8; ++i) acc[i] = vzero8();

  gemm_core_128x128(A16 + (size_t)m0 * K, W16 + (size_t)j0 * K, K,
                    &Al[0][0], &Bl[0][0], acc);

  const int half = (lane >= 16) ? 8 : 0;
#pragma unroll
  for (int jt = 0; jt < 8; ++jt) {
    const int j = j0 + jt * 16 + (lane & 15);
    const float bj = bias[j];
#pragma unroll
    for (int r = 0; r < 8; ++r) {
      const int row = m0 + wv * 16 + half + r;
      const float v = acc[jt][r] + bj;
      if (MODE == 0) {
        outf[(size_t)row * Ncols + j] = v + extra[(size_t)(row % T) * Ncols + j];
      } else if (MODE == 1) {
        outh[(size_t)row * Ncols + j] =
            (_Float16)(0.5f * v * (1.f + erff(v * 0.70710678118654752f)));
      } else {
        outf[(size_t)row * Ncols + j] = v + extra[(size_t)row * Ncols + j];
      }
    }
  }
}

// ---------------------------------------------------------------------------
extern "C" void kernel_launch(void* const* d_in, const int* in_sizes, int n_in,
                              void* d_out, int out_size, void* d_ws, size_t ws_size,
                              hipStream_t stream) {
  const float* feat   = (const float*)d_in[0];
  const int*   bidx   = (const int*)d_in[1];
  const float* query  = (const float*)d_in[3];
  const float* g_q    = (const float*)d_in[4];
  const float* b_q    = (const float*)d_in[5];
  const float* w_q    = (const float*)d_in[6];
  const float* w_k    = (const float*)d_in[7];
  const float* w_v    = (const float*)d_in[8];
  const float* b_q_in = (const float*)d_in[9];
  const float* b_k_in = (const float*)d_in[10];
  const float* b_v_in = (const float*)d_in[11];
  const float* w_o    = (const float*)d_in[12];
  const float* b_o    = (const float*)d_in[13];
  const float* g_ff   = (const float*)d_in[14];
  const float* b_ff   = (const float*)d_in[15];
  const float* w1     = (const float*)d_in[16];
  const float* b1     = (const float*)d_in[17];
  const float* w2     = (const float*)d_in[18];
  const float* b2     = (const float*)d_in[19];
  float* out = (float*)d_out;

  const int N = in_sizes[0] / C;          // 32768
  const int B = out_size / (T * C);       // 16
  const int BT = B * T;

  char* W = (char*)d_ws;
  size_t off = 0;
  auto take = [&](size_t bytes) -> void* {
    void* p = W + off;
    off = (off + bytes + 255) & ~(size_t)255;
    return p;
  };
  float*    q_ln    = (float*)take((size_t)T * C * 4);
  _Float16* qh16    = (_Float16*)take((size_t)H * T * DH * 2);
  _Float16* wk16    = (_Float16*)take((size_t)C * C * 2);
  _Float16* wv16    = (_Float16*)take((size_t)C * C * 2);
  _Float16* wo16    = (_Float16*)take((size_t)C * C * 2);
  _Float16* w116    = (_Float16*)take((size_t)2 * C * C * 2);
  _Float16* w216    = (_Float16*)take((size_t)C * 2 * C * 2);
  _Float16* feat16  = (_Float16*)take((size_t)N * C * 2);
  _Float16* K16     = (_Float16*)take((size_t)N * C * 2);
  _Float16* Vt16    = (_Float16*)take((size_t)C * N * 2);
  int*      segs    = (int*)take(32 * 4);
  _Float16* attnO16 = (_Float16*)take((size_t)BT * C * 2);
  float*    hbuf    = (float*)take((size_t)BT * C * 4);
  _Float16* hn16    = (_Float16*)take((size_t)BT * C * 2);
  _Float16* ff116   = (_Float16*)take((size_t)BT * 2 * C * 2);

  // 1) layernorm(query) + Q projection (tiny)
  ln_rows_kernel<float><<<T, 256, 0, stream>>>(query, g_q, b_q, q_ln);
  qproj_kernel<<<(T * C) / 256, 256, 0, stream>>>(q_ln, w_q, b_q_in, qh16);

  // 2) f16 conversions (feat16 = 32MB -> fits in 192MB L2 for GEMM re-reads)
  f32_to_f16_kernel<<<(int)(((size_t)N * C + 255) / 256), 256, 0, stream>>>(
      feat, feat16, (size_t)N * C);
  f32_to_f16_kernel<<<(C * C + 255) / 256, 256, 0, stream>>>(w_k, wk16, (size_t)C * C);
  f32_to_f16_kernel<<<(C * C + 255) / 256, 256, 0, stream>>>(w_v, wv16, (size_t)C * C);
  f32_to_f16_kernel<<<(C * C + 255) / 256, 256, 0, stream>>>(w_o, wo16, (size_t)C * C);
  f32_to_f16_kernel<<<(2 * C * C + 255) / 256, 256, 0, stream>>>(w1, w116, (size_t)2 * C * C);
  f32_to_f16_kernel<<<(2 * C * C + 255) / 256, 256, 0, stream>>>(w2, w216, (size_t)2 * C * C);

  // 3) K/V projection (34.4 GFLOP): TDM-fed WMMA GEMM
  kvproj_kernel<<<dim3(N / 128, C / 128, 2), 256, 0, stream>>>(
      feat16, wk16, wv16, b_k_in, b_v_in, K16, Vt16, N);

  // 4) per-batch key segments
  seg_init_kernel<<<1, 32, 0, stream>>>(segs);
  seg_bounds_kernel<<<(N + 255) / 256, 256, 0, stream>>>(bidx, segs, N);

  // 5) flash attention (WMMA scores + WMMA P*V)
  attn_kernel<<<dim3(H, B), 64, 0, stream>>>(qh16, K16, Vt16, segs, attnO16, N);

  // 6) epilogue on WMMA: w_o (+query residual), LN, GELU-FFN, down-proj (+h)
  wmma_gemm_kernel<0><<<dim3(BT / 128, C / 128), 256, 0, stream>>>(
      attnO16, wo16, b_o, query, hbuf, nullptr, C, C);
  ln_rows_kernel<_Float16><<<BT, 256, 0, stream>>>(hbuf, g_ff, b_ff, hn16);
  wmma_gemm_kernel<1><<<dim3(BT / 128, (2 * C) / 128), 256, 0, stream>>>(
      hn16, w116, b1, nullptr, nullptr, ff116, C, 2 * C);
  wmma_gemm_kernel<2><<<dim3(BT / 128, C / 128), 256, 0, stream>>>(
      ff116, w216, b2, hbuf, out, nullptr, 2 * C, C);
}
